// CrossBatchEmbeddingMixer_89309549953247
// MI455X (gfx1250) — matrix operand order repaired and checked by reading, not compiled, not run on hardware
//
#include <hip/hip_runtime.h>
#include <math.h>

// CDNA5 / gfx1250 WMMA types
typedef __bf16 v16bf __attribute__((ext_vector_type(16)));
typedef __bf16 v8bf  __attribute__((ext_vector_type(8)));
typedef float  v8f   __attribute__((ext_vector_type(8)));
typedef int    v4i_  __attribute__((vector_size(16)));

#define HDIM 4096
#define BSZ  8192
#define GHID 1024

#define BK   64          // k-tile (two WMMA k-steps)
#define LSTR 72          // padded LDS row stride (bf16 elems), 144B = 16*9

// Async global->LDS path (gfx1250 ASYNCcnt), with graceful fallback.
#if defined(__has_builtin)
#  if __has_builtin(__builtin_amdgcn_global_load_async_to_lds_b128) && \
      __has_builtin(__builtin_amdgcn_s_wait_asynccnt)
#    define USE_ASYNC_LDS 1
#  endif
#endif

#define ASG __attribute__((address_space(1)))
#define ASL __attribute__((address_space(3)))

__device__ __forceinline__ unsigned short f2bf(float f) {
    unsigned int u = __float_as_uint(f);
    u += 0x7FFFu + ((u >> 16) & 1u);           // round to nearest even
    return (unsigned short)(u >> 16);
}

union ABfrag { v16bf v; v8bf h[2]; };

__device__ __forceinline__ void copy16_g2l(const unsigned short* g, unsigned short* l) {
#if defined(USE_ASYNC_LDS)
    __builtin_amdgcn_global_load_async_to_lds_b128(
        (ASG v4i_*)(ASG void*)g,
        (ASL v4i_*)(ASL void*)l, 0, 0);
#else
    *(uint4*)l = *(const uint4*)g;
#endif
}

__device__ __forceinline__ void wait_tile_copies() {
#if defined(USE_ASYNC_LDS)
    __builtin_amdgcn_s_wait_asynccnt(0);
#endif
    __syncthreads();
}

// ---------------------------------------------------------------------------
// f32 -> bf16 conversion (grid-stride)
// ---------------------------------------------------------------------------
__global__ __launch_bounds__(256) void cvt_f32_bf16(const float* __restrict__ in,
                                                    unsigned short* __restrict__ out,
                                                    long long n) {
    long long i = (long long)blockIdx.x * blockDim.x + threadIdx.x;
    long long stride = (long long)gridDim.x * blockDim.x;
    for (; i < n; i += stride) out[i] = f2bf(in[i]);
}

// ---------------------------------------------------------------------------
// Tiled WMMA bf16 GEMM:  C[M,N] = A[M,K] * Bm[N,K]^T   (both K-contiguous, NT)
// Block tile 128x128x64, 256 threads = 8 waves (4x2), wave tile 32x64.
// Double-buffered LDS; tiles staged with GLOBAL_LOAD_ASYNC_TO_LDS_B128 when
// available (ASYNCcnt), else reg-staged copies.
// A source switches to A2 for k >= ksplit (concat-free gate MLP).
// Epilogues:
//   0: Cf f32                     1: Cb bf16
//   2: Cb bf16 transposed (ldct)  3: Cf f32 + Cb bf16
//   4: bf16( gelu(acc + bias[n]) )                      -> Cb
//   5: out = hres + sigmoid(acc + bias[n]) * crossf     (f32 -> outf)
// ---------------------------------------------------------------------------
__global__ __launch_bounds__(256) void gemm_bf16_nt(
    const unsigned short* __restrict__ A,
    const unsigned short* __restrict__ A2, int ksplit, int lda,
    const unsigned short* __restrict__ Bm, int ldb,
    int K,
    float* __restrict__ Cf, unsigned short* __restrict__ Cb,
    int ldc, int ldct,
    const float* __restrict__ bias,
    const float* __restrict__ hres, const float* __restrict__ crossf,
    float* __restrict__ outf,
    int epi)
{
    __shared__ alignas(16) unsigned short As[2][128][LSTR];
    __shared__ alignas(16) unsigned short Bs[2][128][LSTR];

    const int t     = threadIdx.x;
    const int lane  = t & 31;
    const int wave  = t >> 5;
    const int wm    = wave >> 1;          // 0..3  (M direction)
    const int wn    = wave & 1;           // 0..1  (N direction)
    const int m0    = blockIdx.y * 128;
    const int n0    = blockIdx.x * 128;
    const int mlane = lane & 15;
    const int khalf = lane >> 4;

    v8f acc[2][4];
    #pragma unroll
    for (int i = 0; i < 2; ++i)
        #pragma unroll
        for (int j = 0; j < 4; ++j)
            acc[i][j] = (v8f){0.f,0.f,0.f,0.f,0.f,0.f,0.f,0.f};

    // Stage one 128x64 bf16 tile of A and B into LDS buffer `buf`.
    auto issue_tile = [&](int kt, int buf) {
        const int kb = kt * BK;
        const unsigned short* Asrc = A;
        int kc = kb;
        if (kb >= ksplit) { Asrc = A2; kc = kb - ksplit; }
        #pragma unroll
        for (int i = 0; i < 4; ++i) {
            const int q = t + i * 256;          // 1024 16B-chunks per matrix
            const int r = q >> 3;               // 0..127
            const int c = (q & 7) * 8;          // 0..56
            copy16_g2l(Asrc + (size_t)(m0 + r) * lda + kc + c, &As[buf][r][c]);
            copy16_g2l(Bm   + (size_t)(n0 + r) * ldb + kb + c, &Bs[buf][r][c]);
        }
    };

    const int nk = K / BK;
    issue_tile(0, 0);

    for (int kt = 0; kt < nk; ++kt) {
        const int buf = kt & 1;
        wait_tile_copies();                    // asynccnt (if used) + barrier
        if (kt + 1 < nk) issue_tile(kt + 1, buf ^ 1);

        #pragma unroll
        for (int kk = 0; kk < 2; ++kk) {       // two 16x16x32 k-steps per tile
            const int cb = kk * 32;
            ABfrag af[2], bf[4];
            #pragma unroll
            for (int mt = 0; mt < 2; ++mt) {
                const int row = wm * 32 + mt * 16 + mlane;
                af[mt].h[0] = *(const v8bf*)(&As[buf][row][cb + khalf * 8]);
                af[mt].h[1] = *(const v8bf*)(&As[buf][row][cb + 16 + khalf * 8]);
            }
            #pragma unroll
            for (int nt = 0; nt < 4; ++nt) {
                const int row = wn * 64 + nt * 16 + mlane;
                bf[nt].h[0] = *(const v8bf*)(&Bs[buf][row][cb + khalf * 8]);
                bf[nt].h[1] = *(const v8bf*)(&Bs[buf][row][cb + 16 + khalf * 8]);
            }
            #pragma unroll
            for (int mt = 0; mt < 2; ++mt)
                #pragma unroll
                for (int nt = 0; nt < 4; ++nt)
                    acc[mt][nt] = __builtin_amdgcn_wmma_f32_16x16x32_bf16(
                        false, af[mt].v, false, bf[nt].v,
                        (short)0, acc[mt][nt], false, false);
        }
    }

    // ---------------- epilogue ----------------
    #pragma unroll
    for (int mt = 0; mt < 2; ++mt) {
        #pragma unroll
        for (int nt = 0; nt < 4; ++nt) {
            v8f a = acc[mt][nt];
            const int gmb = m0 + wm * 32 + mt * 16 + khalf * 8;   // + v
            const int gn  = n0 + wn * 64 + nt * 16 + mlane;
            if (epi == 0) {
                #pragma unroll
                for (int v = 0; v < 8; ++v)
                    Cf[(size_t)(gmb + v) * ldc + gn] = a[v];
            } else if (epi == 1) {
                #pragma unroll
                for (int v = 0; v < 8; ++v)
                    Cb[(size_t)(gmb + v) * ldc + gn] = f2bf(a[v]);
            } else if (epi == 2) {
                uint4 pk;
                pk.x = (unsigned)f2bf(a[0]) | ((unsigned)f2bf(a[1]) << 16);
                pk.y = (unsigned)f2bf(a[2]) | ((unsigned)f2bf(a[3]) << 16);
                pk.z = (unsigned)f2bf(a[4]) | ((unsigned)f2bf(a[5]) << 16);
                pk.w = (unsigned)f2bf(a[6]) | ((unsigned)f2bf(a[7]) << 16);
                *(uint4*)(Cb + (size_t)gn * ldct + gmb) = pk;     // transposed store
            } else if (epi == 3) {
                #pragma unroll
                for (int v = 0; v < 8; ++v) {
                    const size_t idx = (size_t)(gmb + v) * ldc + gn;
                    Cf[idx] = a[v];
                    Cb[idx] = f2bf(a[v]);
                }
            } else if (epi == 4) {
                const float bv = bias[gn];
                #pragma unroll
                for (int v = 0; v < 8; ++v) {
                    const float x = a[v] + bv;
                    const float g = 0.5f * x * (1.f + erff(x * 0.70710678118654752f));
                    Cb[(size_t)(gmb + v) * ldc + gn] = f2bf(g);
                }
            } else {
                const float bv = bias[gn];
                #pragma unroll
                for (int v = 0; v < 8; ++v) {
                    const float x = a[v] + bv;
                    const float s = 1.f / (1.f + __expf(-x));
                    const size_t idx = (size_t)(gmb + v) * ldc + gn;
                    outf[idx] = hres[idx] + s * crossf[idx];
                }
            }
        }
    }
}

// ---------------------------------------------------------------------------
// L2-normalize rows of proj (f32) -> normed bf16.  One block per row.
// ---------------------------------------------------------------------------
__global__ __launch_bounds__(256) void normalize_rows(const float* __restrict__ proj,
                                                      unsigned short* __restrict__ out) {
    __shared__ float red[256];
    const int row = blockIdx.x, t = threadIdx.x;
    const float* p = proj + (size_t)row * HDIM;
    float s = 0.f;
    for (int j = t; j < HDIM; j += 256) { const float x = p[j]; s += x * x; }
    red[t] = s; __syncthreads();
    for (int o = 128; o > 0; o >>= 1) { if (t < o) red[t] += red[t + o]; __syncthreads(); }
    const float inv = 1.f / fmaxf(sqrtf(red[0]), 1e-12f);
    unsigned short* o = out + (size_t)row * HDIM;
    for (int j = t; j < HDIM; j += 256) o[j] = f2bf(p[j] * inv);
}

// ---------------------------------------------------------------------------
// Masked row softmax over sim (f32) -> weights bf16. Masks diag + ~mask cols.
// Fully-masked rows produce zeros (nan_to_num semantics).
// ---------------------------------------------------------------------------
__global__ __launch_bounds__(256) void softmax_rows(const float* __restrict__ sim,
                                                    const unsigned char* __restrict__ mask,
                                                    unsigned short* __restrict__ wout) {
    __shared__ float red[256];
    const int row = blockIdx.x, t = threadIdx.x;
    const float* p = sim + (size_t)row * BSZ;
    float mx = -3.0e38f;
    for (int j = t; j < BSZ; j += 256)
        if (mask[j] && j != row) mx = fmaxf(mx, p[j]);
    red[t] = mx; __syncthreads();
    for (int o = 128; o > 0; o >>= 1) { if (t < o) red[t] = fmaxf(red[t], red[t + o]); __syncthreads(); }
    mx = red[0];
    const bool anyvalid = (mx > -2.9e38f);
    __syncthreads();
    float s = 0.f;
    if (anyvalid)
        for (int j = t; j < BSZ; j += 256)
            if (mask[j] && j != row) s += __expf(p[j] - mx);
    red[t] = s; __syncthreads();
    for (int o = 128; o > 0; o >>= 1) { if (t < o) red[t] += red[t + o]; __syncthreads(); }
    const float sum = red[0];
    const float inv = (anyvalid && sum > 0.f) ? 1.f / sum : 0.f;
    unsigned short* w = wout + (size_t)row * BSZ;
    for (int j = t; j < BSZ; j += 256) {
        const float v = (anyvalid && mask[j] && j != row) ? __expf(p[j] - mx) * inv : 0.f;
        w[j] = f2bf(v);
    }
}

// ---------------------------------------------------------------------------
extern "C" void kernel_launch(void* const* d_in, const int* in_sizes, int n_in,
                              void* d_out, int out_size, void* d_ws, size_t ws_size,
                              hipStream_t stream) {
    const float* hidden         = (const float*)d_in[0];
    const unsigned char* amask  = (const unsigned char*)d_in[1];   // jnp bool = 1 byte
    const float* W_sim          = (const float*)d_in[2];
    const float* W_val          = (const float*)d_in[3];
    const float* W_g1           = (const float*)d_in[4];
    const float* b_g1           = (const float*)d_in[5];
    const float* W_g2           = (const float*)d_in[6];
    const float* b_g2           = (const float*)d_in[7];
    float* out = (float*)d_out;

    char* ws = (char*)d_ws;
    size_t off = 0;
    auto wsalloc = [&](size_t bytes) -> void* {
        void* p = ws + off;
        off = (off + bytes + 255) & ~(size_t)255;
        return p;
    };
    unsigned short* hid_bf   = (unsigned short*)wsalloc((size_t)BSZ * HDIM * 2);
    unsigned short* wsim_bf  = (unsigned short*)wsalloc((size_t)HDIM * HDIM * 2);
    unsigned short* wval_bf  = (unsigned short*)wsalloc((size_t)HDIM * HDIM * 2);
    unsigned short* wg1_bf   = (unsigned short*)wsalloc((size_t)GHID * 2 * HDIM * 2);
    unsigned short* wg2_bf   = (unsigned short*)wsalloc((size_t)HDIM * GHID * 2);
    float*          proj_f   = (float*)wsalloc((size_t)BSZ * HDIM * 4);
    unsigned short* norm_bf  = (unsigned short*)wsalloc((size_t)BSZ * HDIM * 2);
    unsigned short* valT_bf  = (unsigned short*)wsalloc((size_t)HDIM * BSZ * 2);
    float*          sim_f    = (float*)wsalloc((size_t)BSZ * BSZ * 4);
    unsigned short* wts_bf   = (unsigned short*)wsalloc((size_t)BSZ * BSZ * 2);
    float*          cross_f  = (float*)wsalloc((size_t)BSZ * HDIM * 4);
    unsigned short* cross_bf = (unsigned short*)wsalloc((size_t)BSZ * HDIM * 2);
    unsigned short* h1_bf    = (unsigned short*)wsalloc((size_t)BSZ * GHID * 2);
    (void)ws_size; (void)in_sizes; (void)n_in; (void)out_size;

    const dim3 blk(256);
    const int KSPLIT_NONE = 1 << 30;

    // bf16 copies of all GEMM operands
    cvt_f32_bf16<<<2048, blk, 0, stream>>>(hidden, hid_bf,  (long long)BSZ * HDIM);
    cvt_f32_bf16<<<2048, blk, 0, stream>>>(W_sim,  wsim_bf, (long long)HDIM * HDIM);
    cvt_f32_bf16<<<2048, blk, 0, stream>>>(W_val,  wval_bf, (long long)HDIM * HDIM);
    cvt_f32_bf16<<<2048, blk, 0, stream>>>(W_g1,   wg1_bf,  (long long)GHID * 2 * HDIM);
    cvt_f32_bf16<<<1024, blk, 0, stream>>>(W_g2,   wg2_bf,  (long long)HDIM * GHID);

    // proj = hidden @ W_sim^T  (f32)
    gemm_bf16_nt<<<dim3(HDIM / 128, BSZ / 128), blk, 0, stream>>>(
        hid_bf, hid_bf, KSPLIT_NONE, HDIM, wsim_bf, HDIM, HDIM,
        proj_f, nullptr, HDIM, 0, nullptr, nullptr, nullptr, nullptr, 0);

    normalize_rows<<<BSZ, blk, 0, stream>>>(proj_f, norm_bf);

    // sim = normed @ normed^T  (f32)
    gemm_bf16_nt<<<dim3(BSZ / 128, BSZ / 128), blk, 0, stream>>>(
        norm_bf, norm_bf, KSPLIT_NONE, HDIM, norm_bf, HDIM, HDIM,
        sim_f, nullptr, BSZ, 0, nullptr, nullptr, nullptr, nullptr, 0);

    softmax_rows<<<BSZ, blk, 0, stream>>>(sim_f, amask, wts_bf);

    // values^T = (hidden @ W_val^T)^T  (bf16, transposed store)
    gemm_bf16_nt<<<dim3(HDIM / 128, BSZ / 128), blk, 0, stream>>>(
        hid_bf, hid_bf, KSPLIT_NONE, HDIM, wval_bf, HDIM, HDIM,
        nullptr, valT_bf, 0, BSZ, nullptr, nullptr, nullptr, nullptr, 2);

    // cross = weights @ values  (f32 + bf16)
    gemm_bf16_nt<<<dim3(HDIM / 128, BSZ / 128), blk, 0, stream>>>(
        wts_bf, wts_bf, KSPLIT_NONE, BSZ, valT_bf, BSZ, BSZ,
        cross_f, cross_bf, HDIM, 0, nullptr, nullptr, nullptr, nullptr, 3);

    // h1 = gelu([hidden|cross] @ W_g1^T + b_g1)  (bf16), concat via ksplit
    gemm_bf16_nt<<<dim3(GHID / 128, BSZ / 128), blk, 0, stream>>>(
        hid_bf, cross_bf, HDIM, HDIM, wg1_bf, 2 * HDIM, 2 * HDIM,
        nullptr, h1_bf, GHID, 0, b_g1, nullptr, nullptr, nullptr, 4);

    // out = hidden + sigmoid(h1 @ W_g2^T + b_g2) * cross  (f32)
    gemm_bf16_nt<<<dim3(HDIM / 128, BSZ / 128), blk, 0, stream>>>(
        h1_bf, h1_bf, KSPLIT_NONE, GHID, wg2_bf, GHID, GHID,
        nullptr, nullptr, HDIM, 0, b_g2, hidden, cross_f, out, 5);
}
